// GAT_56650618634655
// MI455X (gfx1250) — matrix operand order repaired and checked by reading, not compiled, not run on hardware
//
#include <hip/hip_runtime.h>

// ---------------- problem constants (from reference) ----------------
#define E_N    100000      // line-graph nodes
#define EL_N   1600000     // line-graph edges (without self loops)
#define EC_N   1700000     // EL_N + E_N self loops
#define NN_N   50000       // original graph nodes (output rows)
#define IN_N   16
#define H1_N   4
#define C1_N   64
#define F1_N   256         // H1*C1
#define C2_N   64
#define OUT_N  16
#define NEG_SL 0.2f
#define BN_EPS 1e-5f

typedef __attribute__((ext_vector_type(16))) _Float16 v16h;
typedef __attribute__((ext_vector_type(8)))  _Float16 v8h;
typedef __attribute__((ext_vector_type(8)))  float    v8f;
typedef __attribute__((ext_vector_type(4)))  float    v4f;
typedef __attribute__((ext_vector_type(4)))  int      v4i;
typedef __attribute__((address_space(1)))    v4i      v4i_g;   // global int4
typedef __attribute__((address_space(3)))    v4i      v4i_l;   // LDS int4

__device__ __forceinline__ float lrelu(float x) { return x > 0.f ? x : NEG_SL * x; }

// float atomic-max via monotonic int/uint mapping (init must be very negative)
__device__ __forceinline__ void atomicMaxF(float* addr, float v) {
  if (v >= 0.f) atomicMax((int*)addr, __float_as_int(v));
  else          atomicMin((unsigned int*)addr, __float_as_uint(v));
}

// edge e -> (src,dst); self loops appended after the EL real edges
__device__ __forceinline__ void edge_sd(const int* __restrict__ ei, int e, int& s, int& d) {
  if (e < EL_N) { s = ei[e]; d = ei[EL_N + e]; }
  else          { s = e - EL_N; d = s; }
}

// 16-byte generic->LDS staging helper; prefers gfx1250 async global->LDS DMA.
__device__ __forceinline__ void stage16(const char* g, char* l) {
#if __has_builtin(__builtin_amdgcn_global_load_async_to_lds_b128) && \
    __has_builtin(__builtin_amdgcn_s_wait_asynccnt)
  __builtin_amdgcn_global_load_async_to_lds_b128(
      (v4i_g*)(void*)g, (v4i_l*)(void*)l, 0, 0);
#else
  *(v8h*)l = *(const v8h*)g;
#endif
}
__device__ __forceinline__ void stage_wait() {
#if __has_builtin(__builtin_amdgcn_global_load_async_to_lds_b128) && \
    __has_builtin(__builtin_amdgcn_s_wait_asynccnt)
  __builtin_amdgcn_s_wait_asynccnt(0);
#endif
}

// ---------------- generic fill ----------------
__global__ void fill_f32(float* __restrict__ p, int n, float v) {
  int t = blockIdx.x * blockDim.x + threadIdx.x;
  if (t < n) p[t] = v;
}

// ---------------- B pre-pack into WMMA lane order (f16) ----------------
// Bp[((ck*ntN + nt)*32 + lane)*16 + i] = B[k, ncol], k = ck*32 + 8*hf + (i&7) + 16*(i/8)
__global__ void pack_b_k(const float* __restrict__ B, _Float16* __restrict__ Bp,
                         int N, int K) {
  int t = blockIdx.x * blockDim.x + threadIdx.x;
  int ntN = N >> 4, KC = (K + 31) >> 5;
  int total = KC * ntN * 512;             // 512 = 32 lanes * 16 halves
  if (t >= total) return;
  int i = t & 15, lane = (t >> 4) & 31, rest = t >> 9;
  int nt = rest % ntN, ck = rest / ntN;
  int hf = lane >> 4, ncol = (nt << 4) + (lane & 15);
  int k = ck * 32 + 8 * hf + (i & 7) + 16 * (i >> 3);
  Bp[t] = (_Float16)((k < K) ? B[(size_t)k * N + ncol] : 0.0f);
}

// ---------------- WMMA GEMM: C[MxNT] = A[MxKT](f32) * Bp(packed f16) -------------
// One wave per (M-tile, group of NG N-tiles): A k-chunk loaded once and reused for
// NG WMMAs. Packed B staged block-wide in LDS (async DMA when available), read
// back with ds_load_b128. All threads reach the barrier; tile clamp keeps EXEC
// all-1s at every WMMA (ISA 7.12 requirement).
template <int KT, int NT, int NG>
__global__ void wmma_gemm_t(const float* __restrict__ A, const _Float16* __restrict__ Bp,
                            float* __restrict__ C, int M) {
  constexpr int ntN = NT >> 4;
  constexpr int KC  = (KT + 31) >> 5;
  constexpr int NGR = ntN / NG;                 // N-tile groups per M-tile
  constexpr int BBYTES = KC * ntN * 512 * 2;    // packed B size
  __shared__ __attribute__((aligned(32))) _Float16 Bsh[KC * ntN * 512];

  // ---- stage packed B into LDS (block-cooperative, 16B per lane-iter) ----
  for (int ofs = threadIdx.x * 16; ofs < BBYTES; ofs += blockDim.x * 16)
    stage16((const char*)Bp + ofs, (char*)Bsh + ofs);
  stage_wait();
  __syncthreads();

  int wave = (int)((blockIdx.x * blockDim.x + threadIdx.x) >> 5);
  int lane = threadIdx.x & 31;
  int nwave = (M >> 4) * NGR;
  if (wave >= nwave) wave = nwave - 1;          // clamp: EXEC stays all-1s
  int mt = wave / NGR, ng = wave % NGR;
  int hf   = lane >> 4;
  int mrow = (mt << 4) + (lane & 15);
  const float* Arow = A + (size_t)mrow * KT;

  v8f acc[NG];
#pragma unroll
  for (int g = 0; g < NG; ++g) acc[g] = (v8f){};

#pragma unroll
  for (int ck = 0; ck < KC; ++ck) {
    const int kk = ck * 32;
    v16h a;
    {
      const v4f* p1 = (const v4f*)(Arow + kk + 8 * hf);
      v4f f0 = p1[0], f1 = p1[1];
#pragma unroll
      for (int q = 0; q < 4; ++q) { a[q] = (_Float16)f0[q]; a[4 + q] = (_Float16)f1[q]; }
      if constexpr ((KT & 31) == 0) {
        const v4f* p2 = (const v4f*)(Arow + kk + 16 + 8 * hf);
        v4f f2 = p2[0], f3 = p2[1];
#pragma unroll
        for (int q = 0; q < 4; ++q) { a[8 + q] = (_Float16)f2[q]; a[12 + q] = (_Float16)f3[q]; }
      } else {            // KT==16: upper k-half is compile-time zero padding
#pragma unroll
        for (int q = 0; q < 8; ++q) a[8 + q] = (_Float16)0.0f;
      }
    }
#pragma unroll
    for (int g = 0; g < NG; ++g) {
      int nt = ng * NG + g;
      v16h b = *(const v16h*)(&Bsh[((ck * ntN + nt) * 32 + lane) * 16]);
      acc[g] = __builtin_amdgcn_wmma_f32_16x16x32_f16(false, a, false, b, (short)0,
                                                      acc[g], false, false);
    }
  }
#pragma unroll
  for (int g = 0; g < NG; ++g) {
    int ncol = ((ng * NG + g) << 4) + (lane & 15);
#pragma unroll
    for (int r = 0; r < 8; ++r) {
      int m = (mt << 4) + r + 8 * hf;
      C[(size_t)m * NT + ncol] = acc[g][r];
    }
  }
}

// ---------------- readout WMMA: out[NN x 16] = mean[NN x 64] @ Wout + bout ----------
__global__ void wmma_readout(const float* __restrict__ sums, const float* __restrict__ cnt,
                             const _Float16* __restrict__ Bp, const float* __restrict__ bout,
                             float* __restrict__ out) {
  constexpr int K = C2_N;               // 64, N = 16 -> single N tile, 2 k-chunks
  int wave = (int)((blockIdx.x * blockDim.x + threadIdx.x) >> 5);
  int lane = threadIdx.x & 31;
  int ntile = NN_N >> 4;
  if (wave >= ntile) wave = ntile - 1;
  int mt = wave;
  int hf   = lane >> 4;
  int mrow = (mt << 4) + (lane & 15);
  int ncol = lane & 15;
  float c   = cnt[mrow];
  float inv = (c > 0.f) ? (1.0f / c) : 0.0f;
  const float* Arow = sums + (size_t)mrow * K;
  v8f acc = {};
#pragma unroll
  for (int ck = 0; ck < 2; ++ck) {
    const int kk = ck * 32;
    v16h a;
    const v4f* p1 = (const v4f*)(Arow + kk + 8 * hf);
    const v4f* p2 = (const v4f*)(Arow + kk + 16 + 8 * hf);
    v4f f0 = p1[0], f1 = p1[1], f2 = p2[0], f3 = p2[1];
#pragma unroll
    for (int q = 0; q < 4; ++q) {
      a[q]      = (_Float16)(f0[q] * inv);
      a[4 + q]  = (_Float16)(f1[q] * inv);
      a[8 + q]  = (_Float16)(f2[q] * inv);
      a[12 + q] = (_Float16)(f3[q] * inv);
    }
    v16h b = *(const v16h*)(Bp + ((size_t)ck * 32 + lane) * 16);
    acc = __builtin_amdgcn_wmma_f32_16x16x32_f16(false, a, false, b, (short)0, acc,
                                                 false, false);
  }
#pragma unroll
  for (int r = 0; r < 8; ++r) {
    int m = (mt << 4) + r + 8 * hf;
    out[(size_t)m * OUT_N + ncol] = acc[r] + bout[ncol];
  }
}

// ---------------- attention logits: al_s/al_d per (node, head) ----------------
__global__ void att_logits(const float* __restrict__ xp, const float* __restrict__ a_s,
                           const float* __restrict__ a_d, float* __restrict__ als,
                           float* __restrict__ ald, int nodes, int heads, int ch) {
  int t = blockIdx.x * blockDim.x + threadIdx.x;
  if (t >= nodes * heads) return;
  int v = t / heads, h = t % heads;
  const float* row = xp + (size_t)v * heads * ch + (size_t)h * ch;
  float s = 0.f, d = 0.f;
  for (int c = 0; c < ch; c += 4) {
    v4f xv = *(const v4f*)(row + c);
    v4f sv = *(const v4f*)(a_s + h * ch + c);
    v4f dv = *(const v4f*)(a_d + h * ch + c);
    s += xv[0] * sv[0] + xv[1] * sv[1] + xv[2] * sv[2] + xv[3] * sv[3];
    d += xv[0] * dv[0] + xv[1] * dv[1] + xv[2] * dv[2] + xv[3] * dv[3];
  }
  als[t] = s; ald[t] = d;
}

// ---------------- edge passes ----------------
__global__ void edge_max_k(const int* __restrict__ ei, const float* __restrict__ als,
                           const float* __restrict__ ald, float* __restrict__ m, int heads) {
  int t = blockIdx.x * blockDim.x + threadIdx.x;
  if (t >= EC_N * heads) return;
  int e = t / heads, h = t % heads, s, d;
  edge_sd(ei, e, s, d);
  atomicMaxF(&m[d * heads + h], lrelu(als[s * heads + h] + ald[d * heads + h]));
}

__global__ void edge_exp_k(const int* __restrict__ ei, const float* __restrict__ als,
                           const float* __restrict__ ald, const float* __restrict__ m,
                           float* __restrict__ den, float* __restrict__ alph, int heads) {
  int t = blockIdx.x * blockDim.x + threadIdx.x;
  if (t >= EC_N * heads) return;
  int e = t / heads, h = t % heads, s, d;
  edge_sd(ei, e, s, d);
  float ex = __expf(lrelu(als[s * heads + h] + ald[d * heads + h]) - m[d * heads + h]);
  alph[t] = ex;
  atomicAdd(&den[d * heads + h], ex);
}

__global__ void edge_div_k(const int* __restrict__ ei, const float* __restrict__ den,
                           float* __restrict__ alph, int heads) {
  int t = blockIdx.x * blockDim.x + threadIdx.x;
  if (t >= EC_N * heads) return;
  int e = t / heads, h = t % heads, s, d;
  edge_sd(ei, e, s, d);
  alph[t] = alph[t] / (den[d * heads + h] + 1e-16f);
}

// out[dst, c..c+3] += alpha[e, head(c)] * xp[src, c..c+3] ; float4 per thread
__global__ void edge_scatter_v4(const int* __restrict__ ei, const float* __restrict__ alph,
                                const float* __restrict__ xp, float* __restrict__ out,
                                int heads, int ch) {
  int F = heads * ch, Q = F >> 2;
  int t = blockIdx.x * blockDim.x + threadIdx.x;
  if (t >= EC_N * Q) return;              // <= 108.8e6 threads
  int e = t / Q, c4 = (t % Q) << 2, h = c4 / ch, s, d;
  edge_sd(ei, e, s, d);
  v4f v = *(const v4f*)(xp + (size_t)s * F + c4);
  float al = alph[e * heads + h];
  float* o = out + (size_t)d * F + c4;
  atomicAdd(o + 0, al * v[0]);
  atomicAdd(o + 1, al * v[1]);
  atomicAdd(o + 2, al * v[2]);
  atomicAdd(o + 3, al * v[3]);
}

// ---------------- batchnorm ----------------
#define BN_ROWS 128
__global__ void bn_stats_k(const float* __restrict__ h, float* __restrict__ stats) {
  int c = threadIdx.x;                    // 256 columns
  int r0 = blockIdx.x * BN_ROWS;
  float s = 0.f, s2 = 0.f;
  for (int r = 0; r < BN_ROWS; ++r) {
    int row = r0 + r;
    if (row < E_N) { float v = h[(size_t)row * F1_N + c]; s += v; s2 += v * v; }
  }
  atomicAdd(&stats[c], s);
  atomicAdd(&stats[F1_N + c], s2);
}

__global__ void bn_apply_k(float* __restrict__ h, const float* __restrict__ stats,
                           const float* __restrict__ gamma, const float* __restrict__ beta) {
  int t = blockIdx.x * blockDim.x + threadIdx.x;
  if (t >= E_N * F1_N) return;
  int c = t & (F1_N - 1);
  float mu  = stats[c] * (1.0f / E_N);
  float var = stats[F1_N + c] * (1.0f / E_N) - mu * mu;
  h[t] = (h[t] - mu) * rsqrtf(var + BN_EPS) * gamma[c] + beta[c];
}

// ---------------- bias + readout accumulation ----------------
__global__ void add_bias_k(float* __restrict__ h, const float* __restrict__ b, int n, int ch) {
  int t = blockIdx.x * blockDim.x + threadIdx.x;
  if (t >= n) return;
  h[t] += b[t % ch];
}

__global__ void readout_accum(const float* __restrict__ h2, const int* __restrict__ erow,
                              float* __restrict__ sums, float* __restrict__ cnt) {
  int t = blockIdx.x * blockDim.x + threadIdx.x;
  if (t >= E_N * C2_N) return;
  int v = t >> 6, c = t & 63;
  int r = erow[v];
  atomicAdd(&sums[(size_t)r * C2_N + c], h2[t]);
  if (c == 0) atomicAdd(&cnt[r], 1.0f);
}

// ---------------- host side ----------------
static inline int cdiv(long long a, long long b) { return (int)((a + b - 1) / b); }

extern "C" void kernel_launch(void* const* d_in, const int* in_sizes, int n_in,
                              void* d_out, int out_size, void* d_ws, size_t ws_size,
                              hipStream_t stream) {
  const float* x      = (const float*)d_in[0];
  const float* W1     = (const float*)d_in[1];
  const float* a_src1 = (const float*)d_in[2];
  const float* a_dst1 = (const float*)d_in[3];
  const float* b1     = (const float*)d_in[4];   // cancels exactly under batchnorm
  const float* gamma1 = (const float*)d_in[5];
  const float* beta1  = (const float*)d_in[6];
  const float* W2     = (const float*)d_in[7];
  const float* a_src2 = (const float*)d_in[8];
  const float* a_dst2 = (const float*)d_in[9];
  const float* b2     = (const float*)d_in[10];
  const float* Wout   = (const float*)d_in[11];
  const float* bout   = (const float*)d_in[12];
  const int*   ei     = (const int*)d_in[13];
  const int*   erow   = (const int*)d_in[14];
  (void)b1; (void)in_sizes; (void)n_in; (void)out_size; (void)ws_size;
  float* out = (float*)d_out;

  // ---- workspace carving (~260 MB) ----
  char* ws = (char*)d_ws;
  size_t off = 0;
  float* XP1   = (float*)(ws + off); off += (size_t)E_N * F1_N * 4;   // 102.4 MB
  float* H1B   = (float*)(ws + off); off += (size_t)E_N * F1_N * 4;   // 102.4 MB
  float* ALS1  = (float*)(ws + off); off += (size_t)E_N * H1_N * 4;
  float* ALD1  = (float*)(ws + off); off += (size_t)E_N * H1_N * 4;
  float* M1    = (float*)(ws + off); off += (size_t)E_N * H1_N * 4;
  float* DEN1  = (float*)(ws + off); off += (size_t)E_N * H1_N * 4;
  float* ALPH1 = (float*)(ws + off); off += (size_t)EC_N * H1_N * 4;  // 27.2 MB
  float* STATS = (float*)(ws + off); off += 2 * F1_N * 4;
  float* ALS2  = (float*)(ws + off); off += (size_t)E_N * 4;
  float* ALD2  = (float*)(ws + off); off += (size_t)E_N * 4;
  float* M2    = (float*)(ws + off); off += (size_t)E_N * 4;
  float* DEN2  = (float*)(ws + off); off += (size_t)E_N * 4;
  float* ALPH2 = (float*)(ws + off); off += (size_t)EC_N * 4;         // 6.8 MB
  float* SUMS  = (float*)(ws + off); off += (size_t)NN_N * C2_N * 4;  // 12.8 MB
  float* CNT   = (float*)(ws + off); off += (size_t)NN_N * 4;
  _Float16* BP1 = (_Float16*)(ws + off); off += 1 * 16 * 512 * 2;     // W1 packed
  _Float16* BP2 = (_Float16*)(ws + off); off += 8 * 4  * 512 * 2;     // W2 packed
  _Float16* BPO = (_Float16*)(ws + off); off += 2 * 1  * 512 * 2;     // Wout packed
  // layer-2 activations reuse the (dead after scatter-1) XP1 region
  float* XP2 = XP1;
  float* H2  = (float*)((char*)XP1 + (size_t)E_N * C2_N * 4);

  const int T = 256;
  const float NINF = -3.3e38f;

  // ---- init (every call; harness does not re-zero) ----
  fill_f32<<<cdiv((long long)E_N * F1_N, T), T, 0, stream>>>(H1B, E_N * F1_N, 0.f);
  fill_f32<<<cdiv(E_N * H1_N, T), T, 0, stream>>>(M1,   E_N * H1_N, NINF);
  fill_f32<<<cdiv(E_N * H1_N, T), T, 0, stream>>>(DEN1, E_N * H1_N, 0.f);
  fill_f32<<<cdiv(2 * F1_N, T), T, 0, stream>>>(STATS, 2 * F1_N, 0.f);
  fill_f32<<<cdiv(E_N, T), T, 0, stream>>>(M2,   E_N, NINF);
  fill_f32<<<cdiv(E_N, T), T, 0, stream>>>(DEN2, E_N, 0.f);
  fill_f32<<<cdiv(NN_N * C2_N, T), T, 0, stream>>>(SUMS, NN_N * C2_N, 0.f);
  fill_f32<<<cdiv(NN_N, T), T, 0, stream>>>(CNT, NN_N, 0.f);

  // ---- pack weights into WMMA lane order ----
  pack_b_k<<<cdiv(1 * 16 * 512, T), T, 0, stream>>>(W1,   BP1, F1_N, IN_N);
  pack_b_k<<<cdiv(8 * 4 * 512, T), T, 0, stream>>>(W2,   BP2, C2_N, F1_N);
  pack_b_k<<<cdiv(2 * 1 * 512, T), T, 0, stream>>>(Wout, BPO, OUT_N, C2_N);

  // ---- layer 1: xp1 = x @ W1 (WMMA), attention logits ----
  {
    long long waves = (E_N / 16) * ((F1_N / 16) / 4);   // NG=4 -> 25000 waves
    wmma_gemm_t<IN_N, F1_N, 4><<<cdiv(waves * 32, T), T, 0, stream>>>(x, BP1, XP1, E_N);
  }
  att_logits<<<cdiv(E_N * H1_N, T), T, 0, stream>>>(XP1, a_src1, a_dst1, ALS1, ALD1,
                                                    E_N, H1_N, C1_N);
  // ---- layer 1 softmax over incoming edges ----
  edge_max_k<<<cdiv((long long)EC_N * H1_N, T), T, 0, stream>>>(ei, ALS1, ALD1, M1, H1_N);
  edge_exp_k<<<cdiv((long long)EC_N * H1_N, T), T, 0, stream>>>(ei, ALS1, ALD1, M1, DEN1,
                                                                ALPH1, H1_N);
  edge_div_k<<<cdiv((long long)EC_N * H1_N, T), T, 0, stream>>>(ei, DEN1, ALPH1, H1_N);
  edge_scatter_v4<<<cdiv((long long)EC_N * (F1_N / 4), T), T, 0, stream>>>(
      ei, ALPH1, XP1, H1B, H1_N, C1_N);
  // ---- batchnorm (b1 is a constant shift -> cancels in (h - mu)) ----
  bn_stats_k<<<cdiv(E_N, BN_ROWS), F1_N, 0, stream>>>(H1B, STATS);
  bn_apply_k<<<cdiv((long long)E_N * F1_N, T), T, 0, stream>>>(H1B, STATS, gamma1, beta1);

  // ---- layer 2: xp2 = h1 @ W2 (WMMA, K=256), logits, softmax, scatter ----
  {
    long long waves = (E_N / 16) * ((C2_N / 16) / 4);   // NG=4 -> 6250 waves
    wmma_gemm_t<F1_N, C2_N, 4><<<cdiv(waves * 32, T), T, 0, stream>>>(H1B, BP2, XP2, E_N);
  }
  fill_f32<<<cdiv((long long)E_N * C2_N, T), T, 0, stream>>>(H2, E_N * C2_N, 0.f);
  att_logits<<<cdiv(E_N, T), T, 0, stream>>>(XP2, a_src2, a_dst2, ALS2, ALD2, E_N, 1, C2_N);
  edge_max_k<<<cdiv(EC_N, T), T, 0, stream>>>(ei, ALS2, ALD2, M2, 1);
  edge_exp_k<<<cdiv(EC_N, T), T, 0, stream>>>(ei, ALS2, ALD2, M2, DEN2, ALPH2, 1);
  edge_div_k<<<cdiv(EC_N, T), T, 0, stream>>>(ei, DEN2, ALPH2, 1);
  edge_scatter_v4<<<cdiv((long long)EC_N * (C2_N / 4), T), T, 0, stream>>>(
      ei, ALPH2, XP2, H2, 1, C2_N);
  add_bias_k<<<cdiv((long long)E_N * C2_N, T), T, 0, stream>>>(H2, b2, E_N * C2_N, C2_N);

  // ---- readout: per-row mean over incident edges, then @ Wout + bout (WMMA) ----
  readout_accum<<<cdiv((long long)E_N * C2_N, T), T, 0, stream>>>(H2, erow, SUMS, CNT);
  {
    int tiles = NN_N / 16;                            // 3125 tiles
    wmma_readout<<<cdiv((long long)tiles * 32, T), T, 0, stream>>>(SUMS, CNT, BPO, bout,
                                                                   out);
  }
}
